// MambaBlock_3822520893651
// MI455X (gfx1250) — matrix operand order repaired and checked by reading, not compiled, not run on hardware
//
#include <hip/hip_runtime.h>

#define BATCH  4
#define SEQ    4096
#define DMODEL 1024
#define DINNER 2048
#define TOK    (BATCH*SEQ)   /* 16384 */
#define NCHUNK (SEQ/16)      /* 256 */
#define KSTEP  32

typedef __attribute__((ext_vector_type(16))) __bf16 v16bf;
typedef __attribute__((ext_vector_type(8)))  __bf16 v8bf;
typedef __attribute__((ext_vector_type(4)))  __bf16 v4bf;
typedef __attribute__((ext_vector_type(8)))  float  v8f;
typedef __attribute__((ext_vector_type(4)))  float  v4f;
typedef int v4i_vs __attribute__((vector_size(16)));   // matches builtin param type

#if __has_builtin(__builtin_amdgcn_global_load_async_to_lds_b128)
#define HAVE_ASYNC_LDS 1
#endif

__device__ __forceinline__ v8f wmma_bf16f32(v16bf a, v16bf b, v8f c) {
  return __builtin_amdgcn_wmma_f32_16x16x32_bf16(false, a, false, b, (short)0, c, false, false);
}

// 16B global -> LDS copy; async (no VGPR round-trip, ASYNCcnt) when available.
__device__ __forceinline__ void copy16_g2l(const __bf16* src, __bf16* dst) {
#ifdef HAVE_ASYNC_LDS
  __builtin_amdgcn_global_load_async_to_lds_b128(
      (__attribute__((address_space(1))) v4i_vs*)src,
      (__attribute__((address_space(3))) v4i_vs*)dst, 0, 0);
#else
  *(v8bf*)dst = *(const v8bf*)src;
#endif
}

__device__ __forceinline__ void async_wait0() {
#ifdef HAVE_ASYNC_LDS
#if __has_builtin(__builtin_amdgcn_s_wait_asynccnt)
  __builtin_amdgcn_s_wait_asynccnt(0);
#else
  asm volatile("s_wait_asynccnt 0x0" ::: "memory");
#endif
#endif
}

// A-operand frag, source row-major bf16 [M][ld]: lane row = lane&15,
// K = kb..kb+15, kb = lane<16 ? 0 : 16.  Two 16B loads
// (global_load_b128 or ds_load_b128 depending on pointer).
__device__ __forceinline__ v16bf load_a_frag(const __bf16* base, int ld, int row0, int k0, int lane) {
  int r  = row0 + (lane & 15);
  int kb = (lane < 16) ? 0 : 16;
  const __bf16* p = base + (size_t)r * ld + k0 + kb;
  v8bf lo = *(const v8bf*)(p);
  v8bf hi = *(const v8bf*)(p + 8);
  return __builtin_shufflevector(lo, hi, 0,1,2,3,4,5,6,7,8,9,10,11,12,13,14,15);
}

// B-operand frag, source stored [N][K] bf16 (K-contiguous): lane col = lane&15,
// K = kb..kb+15, kb = lane<16 ? 0 : 16.  One 32B load.
__device__ __forceinline__ v16bf load_b_frag(const __bf16* baseT, int ldk, int col0, int k0, int lane) {
  int n  = col0 + (lane & 15);
  int kb = (lane < 16) ? 0 : 16;
  return *(const v16bf*)(baseT + (size_t)n * ldk + k0 + kb);
}

// ---------------- conversion kernels ----------------

__global__ void __launch_bounds__(256) conv_bf16_k(const float* __restrict__ in,
                                                   __bf16* __restrict__ out, int n4) {
  int i = blockIdx.x * 256 + threadIdx.x;
  if (i < n4) {
    v4f v = *(const v4f*)(in + (size_t)i * 4);
    v4bf o;
#pragma unroll
    for (int j = 0; j < 4; j++) o[j] = (__bf16)v[j];
    *(v4bf*)(out + (size_t)i * 4) = o;
  }
}

// w f32 [K][N] -> wT bf16 [N][K], LDS-tiled 32x32.
__global__ void __launch_bounds__(256) transpose_bf16_k(const float* __restrict__ w,
                                                        __bf16* __restrict__ wT, int K, int N) {
  __shared__ float tile[32][33];
  int n0 = blockIdx.x * 32, k0 = blockIdx.y * 32;
  int tx = threadIdx.x, ty = threadIdx.y;   // blockDim = (32, 8)
#pragma unroll
  for (int i = 0; i < 4; i++)
    tile[ty + 8 * i][tx] = w[(size_t)(k0 + ty + 8 * i) * N + n0 + tx];
  __syncthreads();
#pragma unroll
  for (int i = 0; i < 4; i++)
    wT[(size_t)(n0 + ty + 8 * i) * K + k0 + tx] = (__bf16)tile[tx][ty + 8 * i];
}

// ---------------- GEMM: D[m][n] = sum_k A[m][k] * B[n][k] (+bias), store D^T ----------------
// A tile (64 x 32) staged in LDS via async global->LDS copies, double-buffered.
// EPI: 0 -> f32 store to Out[n*M+m], bias by m (GEMM3 -> d_out)
//      1 -> bf16 store to Out[n*M+m], bias by n (x_proj^T)
//      2 -> sigmoid, bf16 store to Out[n*M+m], bias by n (gate^T)
// Block: 256 thr = 8 waves (2x4), wave tile 32(M)x64(N), block tile 64x256.
template<int EPI>
__global__ void __launch_bounds__(256) gemm_wmma_k(const __bf16* __restrict__ Am,
                                                   const __bf16* __restrict__ BT,
                                                   const float* __restrict__ bias,
                                                   void* __restrict__ Outv,
                                                   int M, int N, int K) {
  __shared__ __align__(16) __bf16 sA[2][64][KSTEP];
  int tid  = threadIdx.x;
  int lane = tid & 31;
  int wave = tid >> 5;
  int wm = wave >> 2, wn = wave & 3;
  int mblk = blockIdx.x * 64;
  int m0loc = wm * 32;
  int n0 = blockIdx.y * 256 + wn * 64;

  // one 16B async copy per thread stages the whole 64x32 A tile
  auto stage = [&](int buf, int k0) {
    int row = tid >> 2;
    int seg = (tid & 3) * 8;
    copy16_g2l(Am + (size_t)(mblk + row) * K + k0 + seg, &sA[buf][row][seg]);
  };

  v8f acc[2][4];
#pragma unroll
  for (int i = 0; i < 2; i++)
#pragma unroll
    for (int j = 0; j < 4; j++)
#pragma unroll
      for (int r = 0; r < 8; r++) acc[i][j][r] = 0.f;

  stage(0, 0);
  int buf = 0;
  for (int k0 = 0; k0 < K; k0 += KSTEP) {
    async_wait0();       // own stage copies have landed in LDS
    __syncthreads();     // tile visible to all waves; previous readers drained
    if (k0 + KSTEP < K) {
      stage(buf ^ 1, k0 + KSTEP);                                  // overlap next tile
      __builtin_prefetch(BT + (size_t)(n0 + lane) * K + k0 + KSTEP, 0, 1);
    }
    v16bf a0 = load_a_frag(&sA[buf][0][0], KSTEP, m0loc,      0, lane);  // ds_load_b128 x2
    v16bf a1 = load_a_frag(&sA[buf][0][0], KSTEP, m0loc + 16, 0, lane);
#pragma unroll
    for (int j = 0; j < 4; j++) {
      v16bf b = load_b_frag(BT, K, n0 + 16 * j, k0, lane);
      acc[0][j] = wmma_bf16f32(a0, b, acc[0][j]);
      acc[1][j] = wmma_bf16f32(a1, b, acc[1][j]);
    }
    buf ^= 1;
  }

#pragma unroll
  for (int i = 0; i < 2; i++)
#pragma unroll
    for (int j = 0; j < 4; j++) {
      int n = n0 + 16 * j + (lane & 15);
      int mbase = mblk + m0loc + 16 * i + ((lane < 16) ? 0 : 8);
      if (EPI == 0) {            // f32, bias by m, lane-contiguous 32B store
        v8f ov;
#pragma unroll
        for (int r = 0; r < 8; r++) ov[r] = acc[i][j][r] + bias[mbase + r];
        *(v8f*)((float*)Outv + (size_t)n * M + mbase) = ov;
      } else {
        float bv = bias[n];
        v8bf ov;
#pragma unroll
        for (int r = 0; r < 8; r++) {
          float v = acc[i][j][r] + bv;
          if (EPI == 2) v = 1.f / (1.f + __expf(-v));
          ov[r] = (__bf16)v;
        }
        *(v8bf*)((__bf16*)Outv + (size_t)n * M + mbase) = ov;  // 16B store
      }
    }
}

// ---------------- per-channel chunk matrices ----------------
// Mout = [K_toeplitz | W], Mst = [V | A^16], each 16x32 bf16 row-major.
__global__ void __launch_bounds__(128) ssm_precompute_k(const float* __restrict__ A,
                                                        const float* __restrict__ Bs,
                                                        const float* __restrict__ Cs,
                                                        __bf16* __restrict__ mats) {
  __shared__ float sA[4][16][16];
  __shared__ float sv[4][16][16];   // v_d = A^d B
  __shared__ float su[4][17][16];   // u_d = (A^T)^d C
  __shared__ float sP[4][16][16];   // -> A^16
  __shared__ float sC[4][16];
  int wave = threadIdx.x >> 5;
  int lane = threadIdx.x & 31;
  int c = blockIdx.x * 4 + wave;
  int n = lane;
  bool act = (lane < 16);

  if (act) {
#pragma unroll
    for (int m = 0; m < 16; m++) sA[wave][n][m] = A[(size_t)c * 256 + n * 16 + m];
    sv[wave][0][n] = Bs[c * 16 + n];
    su[wave][0][n] = Cs[c * 16 + n];
    sC[wave][n]    = Cs[c * 16 + n];
  }
  __syncthreads();

  for (int d = 1; d < 16; d++) {
    float s1 = 0.f;
    if (act) { for (int m = 0; m < 16; m++) s1 += sA[wave][n][m] * sv[wave][d - 1][m]; }
    __syncthreads();
    if (act) sv[wave][d][n] = s1;
    __syncthreads();
  }
  for (int d = 1; d <= 16; d++) {
    float s1 = 0.f;
    if (act) { for (int m = 0; m < 16; m++) s1 += sA[wave][m][n] * su[wave][d - 1][m]; }
    __syncthreads();
    if (act) su[wave][d][n] = s1;
    __syncthreads();
  }
  if (act) { for (int m = 0; m < 16; m++) sP[wave][n][m] = sA[wave][n][m]; }
  __syncthreads();
  for (int it = 0; it < 4; it++) {
    float row[16];
    if (act) {
      for (int m = 0; m < 16; m++) {
        float s1 = 0.f;
        for (int j = 0; j < 16; j++) s1 += sP[wave][n][j] * sP[wave][j][m];
        row[m] = s1;
      }
    }
    __syncthreads();
    if (act) { for (int m = 0; m < 16; m++) sP[wave][n][m] = row[m]; }
    __syncthreads();
  }

  if (act) {
    __bf16* mo = mats + (size_t)c * 1024;
    __bf16* ms = mo + 512;
    for (int i = 0; i < 16; i++) {
      float kv = 0.f;
      if (n >= i) { int d = n - i; for (int m = 0; m < 16; m++) kv += sC[wave][m] * sv[wave][d][m]; }
      mo[n * 32 + i] = (__bf16)kv;                  // K[t,i] = C . A^{t-i} B
      ms[n * 32 + i] = (__bf16)sv[wave][15 - i][n]; // V[:,i] = A^{15-i} B
    }
    for (int m = 0; m < 16; m++) {
      mo[n * 32 + 16 + m] = (__bf16)su[wave][n + 1][m]; // W[t] = (A^{t+1})^T C
      ms[n * 32 + 16 + m] = (__bf16)sP[wave][n][m];     // A^16
    }
  }
}

// ---------------- chunked WMMA scan ----------------
// xp_t/gate_t are [C][TOK] (= [C][b*SEQ+s]) -> contiguous vector loads.
// y (gated) written to ybuf [TOK][C] for GEMM3's B operand.
__global__ void __launch_bounds__(256) ssm_scan_k(const __bf16* __restrict__ mats,
                                                  const __bf16* __restrict__ xp_t,
                                                  const __bf16* __restrict__ gate_t,
                                                  __bf16* __restrict__ ybuf) {
  int lane = threadIdx.x & 31;
  int wave = threadIdx.x >> 5;
  int c = blockIdx.x * 8 + wave;
  const __bf16* mo = mats + (size_t)c * 1024;
  v16bf amo = load_a_frag(mo,       32, 0, 0, lane);
  v16bf ams = load_a_frag(mo + 512, 32, 0, 0, lane);
  v8f h;
#pragma unroll
  for (int r = 0; r < 8; r++) h[r] = 0.f;
  int b = lane & 15;
  const __bf16* xrow = xp_t + (size_t)c * TOK + (size_t)(b & 3) * SEQ;
  const __bf16* grow = gate_t + (size_t)c * TOK + (size_t)(b & 3) * SEQ;

  for (int ch = 0; ch < NCHUNK; ch++) {
    int s0 = ch * 16;
    float hlo[8];
#pragma unroll
    for (int r = 0; r < 8; r++) hlo[r] = __shfl(h[r], b, 32);

    v16bf z;
    if (lane < 16) {               // K=0..15 : X chunk, column = batch
      if (b < BATCH) {
        z = *(const v16bf*)(xrow + s0);            // 32B contiguous
      } else {
#pragma unroll
        for (int j = 0; j < 16; j++) z[j] = (__bf16)0.f;
      }
    } else {                       // K=16..31 : previous state h
#pragma unroll
      for (int r = 0; r < 8; r++) { z[r] = (__bf16)hlo[r]; z[r + 8] = (__bf16)h[r]; }
    }

    v8f zero;
#pragma unroll
    for (int r = 0; r < 8; r++) zero[r] = 0.f;
    v8f y = wmma_bf16f32(amo, z, zero);  // y  = K@X + W@h
    h     = wmma_bf16f32(ams, z, zero);  // h' = V@X + A^16@h

    if (b < BATCH) {
      int tb = (lane < 16) ? 0 : 8;
      v8bf gv = *(const v8bf*)(grow + s0 + tb);     // 16B contiguous
#pragma unroll
      for (int r = 0; r < 8; r++) {
        size_t tok = (size_t)b * SEQ + s0 + r + tb;
        ybuf[tok * DINNER + c] = (__bf16)(y[r] * (float)gv[r]);
      }
    }
  }
}

extern "C" void kernel_launch(void* const* d_in, const int* in_sizes, int n_in,
                              void* d_out, int out_size, void* d_ws, size_t ws_size,
                              hipStream_t stream) {
  const float* x      = (const float*)d_in[0];
  const float* w_in   = (const float*)d_in[1];
  const float* b_in   = (const float*)d_in[2];
  const float* w_gate = (const float*)d_in[3];
  const float* b_gate = (const float*)d_in[4];
  const float* A      = (const float*)d_in[5];
  const float* Bs     = (const float*)d_in[6];
  const float* Cs     = (const float*)d_in[7];
  const float* w_out  = (const float*)d_in[8];
  const float* b_out  = (const float*)d_in[9];

  __bf16* p = (__bf16*)d_ws;
  __bf16* xbf     = p; p += (size_t)TOK * DMODEL;      // x, bf16 [TOK][DMODEL]
  __bf16* w_inT   = p; p += (size_t)DINNER * DMODEL;   // [N][K]
  __bf16* w_gateT = p; p += (size_t)DINNER * DMODEL;
  __bf16* w_outT  = p; p += (size_t)DMODEL * DINNER;
  __bf16* xp_t    = p; p += (size_t)DINNER * TOK;      // x_proj^T  [C][TOK]
  __bf16* gate_t  = p; p += (size_t)DINNER * TOK;      // gate^T    [C][TOK]
  __bf16* ybuf    = p; p += (size_t)TOK * DINNER;      // gated y   [TOK][C]
  __bf16* mats    = p;                                 // 2048 x 1024

  // 0) conversions / weight transposes (one-time, tiny)
  conv_bf16_k<<<(TOK * DMODEL / 4 + 255) / 256, 256, 0, stream>>>(x, xbf, TOK * DMODEL / 4);
  transpose_bf16_k<<<dim3(DINNER / 32, DMODEL / 32), dim3(32, 8), 0, stream>>>(w_in,   w_inT,   DMODEL, DINNER);
  transpose_bf16_k<<<dim3(DINNER / 32, DMODEL / 32), dim3(32, 8), 0, stream>>>(w_gate, w_gateT, DMODEL, DINNER);
  transpose_bf16_k<<<dim3(DMODEL / 32, DINNER / 32), dim3(32, 8), 0, stream>>>(w_out,  w_outT,  DINNER, DMODEL);

  // 1) x_proj^T = (x @ w_in + b_in)^T        M=TOK, N=DINNER, K=DMODEL
  gemm_wmma_k<1><<<dim3(TOK / 64, DINNER / 256), 256, 0, stream>>>(
      xbf, w_inT, b_in, (void*)xp_t, TOK, DINNER, DMODEL);
  // 2) gate^T = sigmoid(x @ w_gate + b_gate)^T
  gemm_wmma_k<2><<<dim3(TOK / 64, DINNER / 256), 256, 0, stream>>>(
      xbf, w_gateT, b_gate, (void*)gate_t, TOK, DINNER, DMODEL);
  // 3) per-channel chunk matrices
  ssm_precompute_k<<<DINNER / 4, 128, 0, stream>>>(A, Bs, Cs, mats);
  // 4) chunked WMMA scan
  ssm_scan_k<<<DINNER / 8, 256, 0, stream>>>(mats, xp_t, gate_t, ybuf);
  // 5) out = y @ w_out + b_out   computed as out^T = w_out^T @ y^T:
  //    M=DMODEL, N=TOK, K=DINNER; A=w_outT, B=ybuf; stores straight into d_out[TOK][DMODEL]
  gemm_wmma_k<0><<<dim3(DMODEL / 64, TOK / 256), 256, 0, stream>>>(
      w_outT, ybuf, b_out, d_out, DMODEL, TOK, DINNER);
}
